// GCN_36292473651753
// MI455X (gfx1250) — matrix operand order repaired and checked by reading, not compiled, not run on hardware
//
#include <hip/hip_runtime.h>

// ---------------- problem constants (match reference) ----------------
#define NN     50000
#define NE     800000
#define D_IN   256
#define D_HID  256
#define D_OUT  128
#define BN_EPSF 1e-5f

typedef __attribute__((ext_vector_type(16))) __bf16 v16bf;
typedef __attribute__((ext_vector_type(8)))  float  v8f;

__device__ __forceinline__ unsigned short f2bf(float f) {
    union { float f; unsigned u; } v; v.f = f;
    unsigned u = v.u;
    unsigned r = u + 0x7FFFu + ((u >> 16) & 1u);   // round-to-nearest-even
    return (unsigned short)(r >> 16);
}

// ---------------- utility kernels ----------------
__global__ void k_fill(float* p, float val, int n) {
    int i = blockIdx.x * blockDim.x + threadIdx.x;
    if (i < n) p[i] = val;
}

__global__ void k_deg(const long long* __restrict__ ei, float* __restrict__ deg) {
    int e = blockIdx.x * blockDim.x + threadIdx.x;
    if (e < NE) {
        int dst = (int)ei[NE + e];
        atomicAdd(&deg[dst], 1.0f);
    }
}

__global__ void k_rsqrt(float* p, int n) {
    int i = blockIdx.x * blockDim.x + threadIdx.x;
    if (i < n) p[i] = rsqrtf(p[i]);
}

__global__ void k_f32_to_bf16(const float* __restrict__ in,
                              unsigned short* __restrict__ out, long long n) {
    long long i = (long long)blockIdx.x * blockDim.x + threadIdx.x;
    if (i < n) out[i] = f2bf(in[i]);
}

// Wt[o*K + k] = bf16(W[k*Kout + o])  (transpose so B-fragment loads are contiguous)
__global__ void k_transpose_w(const float* __restrict__ W,
                              unsigned short* __restrict__ Wt, int K, int Kout) {
    int i = blockIdx.x * blockDim.x + threadIdx.x;
    if (i < K * Kout) {
        int o = i / K, k = i % K;
        Wt[i] = f2bf(W[(size_t)k * Kout + o]);
    }
}

// ---------------- WMMA GEMM: C[NN x Kout] = A(bf16)[NN x K] @ Wt(bf16)[Kout x K]^T
// One wave per 16x16 output tile; K-loop in steps of 32 (v_wmma_f32_16x16x32_bf16).
__global__ void k_gemm_wmma(const unsigned short* __restrict__ A,
                            const unsigned short* __restrict__ Wt,
                            float* __restrict__ C, int K, int Kout) {
    int wavesPerBlock = blockDim.x >> 5;
    int wave = blockIdx.x * wavesPerBlock + (threadIdx.x >> 5);
    int lane = threadIdx.x & 31;
    int tiles_n = Kout >> 4;
    int tm = wave / tiles_n;
    int tn = wave % tiles_n;
    if (tm >= (NN / 16)) return;            // wave-uniform guard (EXEC stays all-ones)
    int hf  = lane >> 4;                    // lane half: 0 or 1
    int l16 = lane & 15;

    const unsigned short* aptr = A  + (size_t)(tm * 16 + l16) * K;
    const unsigned short* bptr = Wt + (size_t)(tn * 16 + l16) * K;

    union { v16bf v; unsigned u[8]; } a, b;
    v8f acc = {0.f, 0.f, 0.f, 0.f, 0.f, 0.f, 0.f, 0.f};

    for (int k0 = 0; k0 < K; k0 += 32) {
        __builtin_prefetch(aptr + k0 + 32, 0, 1);   // global_prefetch_b8
        __builtin_prefetch(bptr + k0 + 32, 0, 1);
        // ISA 7.12.2 16-bit A layout: lanes 0-15 -> K[0:8)+[16:24), lanes 16-31 -> K[8:16)+[24:32)
        // 16-bit B layout (col = l16): same K order as A
        #pragma unroll
        for (int v = 0; v < 8; ++v) {
            int j  = 2 * v;
            int ka = (j < 8) ? (hf * 8 + j) : (16 + hf * 8 + (j - 8));
            int kb = ka;
            a.u[v] = *(const unsigned int*)(aptr + k0 + ka);
            b.u[v] = *(const unsigned int*)(bptr + k0 + kb);
        }
        acc = __builtin_amdgcn_wmma_f32_16x16x32_bf16(
                  false, a.v, false, b.v, (short)0, acc, false, false);
    }
    // f32 C/D layout: VGPR r -> row = r + 8*hf, col = l16
    #pragma unroll
    for (int r = 0; r < 8; ++r) {
        int m = tm * 16 + hf * 8 + r;
        C[(size_t)m * Kout + tn * 16 + l16] = acc[r];
    }
}

// agg[i,f] = h[i,f] * dinv[i]^2 + b[f]   (self-loop term + bias)
__global__ void k_self_bias(const float* __restrict__ h, const float* __restrict__ dinv,
                            const float* __restrict__ b, float* __restrict__ agg, int K) {
    long long idx = (long long)blockIdx.x * blockDim.x + threadIdx.x;
    if (idx < (long long)NN * K) {
        int i = (int)(idx / K), f = (int)(idx % K);
        float d = dinv[i];
        agg[idx] = h[idx] * d * d + b[f];
    }
}

// scatter-add edge contributions: agg[dst] += h[src] * dinv[src]*dinv[dst]
__global__ void k_edge_agg(const long long* __restrict__ ei, const float* __restrict__ h,
                           const float* __restrict__ dinv, float* __restrict__ agg, int K) {
    long long t = (long long)blockIdx.x * blockDim.x + threadIdx.x;
    int chunks = K >> 2;
    if (t < (long long)NE * chunks) {
        int e = (int)(t / chunks);
        int c = (int)(t % chunks) * 4;
        int src = (int)ei[e];
        int dst = (int)ei[NE + e];
        float nrm = dinv[src] * dinv[dst];
        float4 v = *(const float4*)(h + (size_t)src * K + c);
        float* ap = agg + (size_t)dst * K + c;
        atomicAdd(ap + 0, v.x * nrm);
        atomicAdd(ap + 1, v.y * nrm);
        atomicAdd(ap + 2, v.z * nrm);
        atomicAdd(ap + 3, v.w * nrm);
    }
}

// per-column sum / sumsq (block = 256 cols, 128 rows per block)
__global__ void k_colstats(const float* __restrict__ hm, float* __restrict__ stats) {
    int c  = threadIdx.x;                 // 0..255 column
    int r0 = blockIdx.x * 128;
    int r1 = r0 + 128; if (r1 > NN) r1 = NN;
    float s = 0.f, q = 0.f;
    for (int r = r0; r < r1; ++r) {
        float v = hm[(size_t)r * D_HID + c];
        s += v; q += v * v;
    }
    atomicAdd(&stats[c], s);
    atomicAdd(&stats[D_HID + c], q);
}

// BN (biased var) + ReLU, emit bf16 operand for next GEMM
__global__ void k_bn_relu_bf16(const float* __restrict__ hm, const float* __restrict__ stats,
                               const float* __restrict__ g, const float* __restrict__ be,
                               unsigned short* __restrict__ xb) {
    long long idx = (long long)blockIdx.x * blockDim.x + threadIdx.x;
    if (idx < (long long)NN * D_HID) {
        int c = (int)(idx & (D_HID - 1));
        const float invN = 1.0f / (float)NN;
        float m   = stats[c] * invN;
        float var = stats[D_HID + c] * invN - m * m;
        float v   = (hm[idx] - m) * rsqrtf(var + BN_EPSF) * g[c] + be[c];
        xb[idx] = f2bf(fmaxf(v, 0.0f));
    }
}

// row-wise log_softmax, one block (128 threads) per row
__global__ void k_log_softmax(const float* __restrict__ z, float* __restrict__ out) {
    __shared__ float red[D_OUT];
    int row = blockIdx.x, c = threadIdx.x;
    float v = z[(size_t)row * D_OUT + c];
    red[c] = v; __syncthreads();
    for (int s = D_OUT / 2; s > 0; s >>= 1) {
        if (c < s) red[c] = fmaxf(red[c], red[c + s]);
        __syncthreads();
    }
    float m = red[0]; __syncthreads();
    red[c] = expf(v - m); __syncthreads();
    for (int s = D_OUT / 2; s > 0; s >>= 1) {
        if (c < s) red[c] += red[c + s];
        __syncthreads();
    }
    float lse = logf(red[0]) + m;
    out[(size_t)row * D_OUT + c] = v - lse;
}

// ---------------- launch ----------------
extern "C" void kernel_launch(void* const* d_in, const int* in_sizes, int n_in,
                              void* d_out, int out_size, void* d_ws, size_t ws_size,
                              hipStream_t stream) {
    const float*     x   = (const float*)d_in[1];
    const long long* ei  = (const long long*)d_in[2];
    const float* W0  = (const float*)d_in[3];
    const float* b0  = (const float*)d_in[4];
    const float* g0  = (const float*)d_in[5];
    const float* be0 = (const float*)d_in[6];
    const float* W1  = (const float*)d_in[7];
    const float* b1  = (const float*)d_in[8];
    const float* g1  = (const float*)d_in[9];
    const float* be1 = (const float*)d_in[10];
    const float* W2  = (const float*)d_in[11];
    const float* b2  = (const float*)d_in[12];
    float* out = (float*)d_out;

    auto au = [](size_t v) { return (v + 255) & ~(size_t)255; };
    char* p = (char*)d_ws;
    float*          dinv  = (float*)p;          p += au(sizeof(float) * NN);
    unsigned short* xb    = (unsigned short*)p; p += au((size_t)2 * NN * D_HID);
    float*          h     = (float*)p;          p += au((size_t)4 * NN * D_HID);
    float*          agg   = (float*)p;          p += au((size_t)4 * NN * D_HID);
    unsigned short* wt    = (unsigned short*)p; p += au((size_t)2 * D_HID * D_HID);
    float*          stats = (float*)p;          p += au(sizeof(float) * 2 * D_HID);

    const int B = 256;
    auto cdiv = [](long long a, long long b) { return (int)((a + b - 1) / b); };

    // degrees: dinv = rsqrt(1 + indeg)
    k_fill <<<cdiv(NN, B), B, 0, stream>>>(dinv, 1.0f, NN);
    k_deg  <<<cdiv(NE, B), B, 0, stream>>>(ei, dinv);
    k_rsqrt<<<cdiv(NN, B), B, 0, stream>>>(dinv, NN);

    // x -> bf16
    k_f32_to_bf16<<<cdiv((long long)NN * D_IN, B), B, 0, stream>>>(x, xb, (long long)NN * D_IN);

    // -------- layer 0 --------
    k_transpose_w<<<cdiv(D_IN * D_HID, B), B, 0, stream>>>(W0, wt, D_IN, D_HID);
    k_gemm_wmma<<<((NN / 16) * (D_HID / 16)) / 8, 256, 0, stream>>>(xb, wt, h, D_IN, D_HID);
    k_self_bias<<<cdiv((long long)NN * D_HID, B), B, 0, stream>>>(h, dinv, b0, agg, D_HID);
    k_edge_agg <<<cdiv((long long)NE * (D_HID / 4), B), B, 0, stream>>>(ei, h, dinv, agg, D_HID);
    k_fill     <<<cdiv(2 * D_HID, B), B, 0, stream>>>(stats, 0.0f, 2 * D_HID);
    k_colstats <<<cdiv(NN, 128), 256, 0, stream>>>(agg, stats);
    k_bn_relu_bf16<<<cdiv((long long)NN * D_HID, B), B, 0, stream>>>(agg, stats, g0, be0, xb);

    // -------- layer 1 --------
    k_transpose_w<<<cdiv(D_HID * D_HID, B), B, 0, stream>>>(W1, wt, D_HID, D_HID);
    k_gemm_wmma<<<((NN / 16) * (D_HID / 16)) / 8, 256, 0, stream>>>(xb, wt, h, D_HID, D_HID);
    k_self_bias<<<cdiv((long long)NN * D_HID, B), B, 0, stream>>>(h, dinv, b1, agg, D_HID);
    k_edge_agg <<<cdiv((long long)NE * (D_HID / 4), B), B, 0, stream>>>(ei, h, dinv, agg, D_HID);
    k_fill     <<<cdiv(2 * D_HID, B), B, 0, stream>>>(stats, 0.0f, 2 * D_HID);
    k_colstats <<<cdiv(NN, 128), 256, 0, stream>>>(agg, stats);
    k_bn_relu_bf16<<<cdiv((long long)NN * D_HID, B), B, 0, stream>>>(agg, stats, g1, be1, xb);

    // -------- layer 2 (out) --------
    k_transpose_w<<<cdiv(D_HID * D_OUT, B), B, 0, stream>>>(W2, wt, D_HID, D_OUT);
    k_gemm_wmma<<<((NN / 16) * (D_OUT / 16)) / 8, 256, 0, stream>>>(xb, wt, h, D_HID, D_OUT);
    k_self_bias<<<cdiv((long long)NN * D_OUT, B), B, 0, stream>>>(h, dinv, b2, agg, D_OUT);
    k_edge_agg <<<cdiv((long long)NE * (D_OUT / 4), B), B, 0, stream>>>(ei, h, dinv, agg, D_OUT);
    k_log_softmax<<<NN, D_OUT, 0, stream>>>(agg, out);
}
